// NodeToEdge_37048387895510
// MI455X (gfx1250) — matrix operand order repaired and checked by reading, not compiled, not run on hardware
//
#include <hip/hip_runtime.h>
#include <hip/hip_bf16.h>

typedef __attribute__((ext_vector_type(16))) _Float16 v16h;
typedef __attribute__((ext_vector_type(8)))  _Float16 v8h;
typedef __attribute__((ext_vector_type(8)))  float    v8f;
typedef __attribute__((ext_vector_type(4)))  float    v4f;

#define NN   512   // nodes
#define DD   256   // node dim
#define HH   256   // hidden (per q/k)
#define EE   128   // edge dim
#define LN_EPS_F 1e-5f

// ---------------------------------------------------------------------------
// Kernel 1: per-row LayerNorm + projection -> q (f16), k (f32),
//           Qd[j,e] = q[j]·w_d[e] + o_b[e],  Kd[i,e] = k[i]·w_d[e]
// One block per node row, 256 threads (8 wave32).
// ---------------------------------------------------------------------------
__global__ __launch_bounds__(256) void ln_proj_kernel(
    const float* __restrict__ node, const float* __restrict__ ln_w,
    const float* __restrict__ ln_b, const float* __restrict__ proj_w,
    const float* __restrict__ proj_b, const float* __restrict__ o_w,
    const float* __restrict__ o_b,
    _Float16* __restrict__ qf16, float* __restrict__ kf32,
    float* __restrict__ qdp, float* __restrict__ kdm)
{
    __shared__ float s_red[16];
    __shared__ float s_ln[DD];
    __shared__ float s_q[HH];
    __shared__ float s_k[HH];

    const int row = blockIdx.x;
    const int t   = threadIdx.x;           // 0..255, one element of the row

    float x  = node[row * DD + t];
    float v1 = x, v2 = x * x;
    #pragma unroll
    for (int off = 16; off > 0; off >>= 1) {
        v1 += __shfl_down(v1, off, 32);
        v2 += __shfl_down(v2, off, 32);
    }
    const int wid = t >> 5, lane = t & 31;
    if (lane == 0) { s_red[wid] = v1; s_red[8 + wid] = v2; }
    __syncthreads();
    float s1 = 0.f, s2 = 0.f;
    #pragma unroll
    for (int w = 0; w < 8; ++w) { s1 += s_red[w]; s2 += s_red[8 + w]; }
    const float mu  = s1 * (1.0f / DD);
    const float var = s2 * (1.0f / DD) - mu * mu;
    const float sn  = (x - mu) * __frsqrt_rn(var + LN_EPS_F) * ln_w[t] + ln_b[t];
    s_ln[t] = sn;
    __syncthreads();

    // projection: out[c] = sum_h s_ln[h] * proj_w[c*DD + h] + proj_b[c]
    float aq = proj_b[t];
    float ak = proj_b[t + HH];
    const float* wq = proj_w + (size_t)t * DD;
    const float* wk = proj_w + (size_t)(t + HH) * DD;
    #pragma unroll 4
    for (int h = 0; h < DD; ++h) {
        const float s = s_ln[h];
        aq = __builtin_fmaf(s, wq[h], aq);
        ak = __builtin_fmaf(s, wk[h], ak);
    }
    qf16[row * HH + t] = (_Float16)aq;
    kf32[row * HH + t] = ak;
    s_q[t] = aq;
    s_k[t] = ak;
    __syncthreads();

    // Qd (+o_b folded) and Kd: w_d[e][h] = o_w[e*(2H) + H + h]
    if (t < EE) {
        const int e = t;
        float acc = o_b[e];
        const float* wd = o_w + (size_t)e * (2 * HH) + HH;
        #pragma unroll 4
        for (int h = 0; h < HH; ++h) acc = __builtin_fmaf(s_q[h], wd[h], acc);
        qdp[row * EE + e] = acc;
    } else {
        const int e = t - EE;
        float acc = 0.f;
        const float* wd = o_w + (size_t)e * (2 * HH) + HH;
        #pragma unroll 4
        for (int h = 0; h < HH; ++h) acc = __builtin_fmaf(s_k[h], wd[h], acc);
        kdm[row * EE + e] = acc;
    }
}

// ---------------------------------------------------------------------------
// Kernel 2: per-i WMMA GEMM.  out[i,j,e] = sum_h A_i[e,h]*q[j,h] + Qd[j,e]-Kd[i,e]
// A_i[e,h] = k[i,h]*w_p[e,h] built in LDS (f16).  One block per i, 8 waves,
// wave w owns e-tile 16w, iterating 32 j-tiles of 16, K = 256 in 8 WMMA steps.
// ---------------------------------------------------------------------------
__global__ __launch_bounds__(256) void edge_wmma_kernel(
    const _Float16* __restrict__ qf16, const float* __restrict__ kf32,
    const float* __restrict__ o_w, const float* __restrict__ qdp,
    const float* __restrict__ kdm, float* __restrict__ out)
{
    extern __shared__ _Float16 A[];        // [EE][HH] halves = 64 KB

    const int i = blockIdx.x;
    const int t = threadIdx.x;

    // Build A_i: thread t owns column h=t for every e (coalesced everywhere).
    {
        const float kv = kf32[i * HH + t];
        const float* wp = o_w + t;         // w_p[e][t] = o_w[e*(2H) + t]
        #pragma unroll 4
        for (int e = 0; e < EE; ++e)
            A[e * HH + t] = (_Float16)(kv * wp[(size_t)e * (2 * HH)]);
    }
    __syncthreads();

    const int wv  = t >> 5;                // wave id 0..7 -> e-tile
    const int ln  = t & 31;
    const int l15 = ln & 15;
    const int hi  = ln >> 4;
    const int e0  = wv * 16;

    // A fragments: ISA 16-bit A layout. lane L: row M = e0 + (L&15),
    // element u (0..15): h = 32*ks + 16*(u>>3) + 8*(L>>4) + (u&7).
    v16h afrag[8];
    {
        const _Float16* arow = A + (e0 + l15) * HH + hi * 8;
        #pragma unroll
        for (int ks = 0; ks < 8; ++ks) {
            union { v16h v; v8h h[2]; } u;
            u.h[0] = *(const v8h*)(arow + ks * 32);
            u.h[1] = *(const v8h*)(arow + ks * 32 + 16);
            afrag[ks] = u.v;
        }
    }

    // Kd bias for this wave's 8 contiguous e's per lane: e = e0 + 8*hi + 0..7
    const float* kdp = kdm + i * EE + e0 + hi * 8;
    const v4f kd0 = *(const v4f*)(kdp);
    const v4f kd1 = *(const v4f*)(kdp + 4);

    float* outbase = out + (size_t)i * NN * EE;

    for (int jt = 0; jt < 32; ++jt) {
        const int j = jt * 16 + l15;
        // B fragment: lane L -> column N = L&15 (row j of q), halves
        // h = 32*ks + 16*(u>>3) + 8*(L>>4) + (u&7), u=0..15.
        const _Float16* qrow = qf16 + j * HH + hi * 8;

        v8f acc = {};
        #pragma unroll
        for (int ks = 0; ks < 8; ++ks) {
            union { v16h v; v8h h[2]; } bu;
            bu.h[0] = *(const v8h*)(qrow + ks * 32);
            bu.h[1] = *(const v8h*)(qrow + ks * 32 + 16);
            const v16h b = bu.v;
            acc = __builtin_amdgcn_wmma_f32_16x16x32_f16(
                false, afrag[ks], false, b, (short)0, acc, false, false);
        }

        // Epilogue: D element (lane L, vgpr v) = (e0 + 8*hi + v, j0 + (L&15))
        // -> each lane's 8 accs are contiguous in e.
        const float* qd = qdp + j * EE + e0 + hi * 8;
        const v4f q0 = *(const v4f*)(qd);
        const v4f q1 = *(const v4f*)(qd + 4);
        v4f r0, r1;
        #pragma unroll
        for (int c = 0; c < 4; ++c) {
            r0[c] = acc[c]     + q0[c] - kd0[c];
            r1[c] = acc[c + 4] + q1[c] - kd1[c];
        }
        float* op = outbase + (size_t)j * EE + e0 + hi * 8;
        __builtin_nontemporal_store(r0, (v4f*)op);
        __builtin_nontemporal_store(r1, (v4f*)(op + 4));
    }
}

// ---------------------------------------------------------------------------
extern "C" void kernel_launch(void* const* d_in, const int* in_sizes, int n_in,
                              void* d_out, int out_size, void* d_ws, size_t ws_size,
                              hipStream_t stream) {
    const float* node   = (const float*)d_in[0];
    const float* ln_w   = (const float*)d_in[1];
    const float* ln_b   = (const float*)d_in[2];
    const float* proj_w = (const float*)d_in[3];
    const float* proj_b = (const float*)d_in[4];
    const float* o_w    = (const float*)d_in[5];
    const float* o_b    = (const float*)d_in[6];
    float* out = (float*)d_out;

    unsigned char* ws = (unsigned char*)d_ws;
    _Float16* qf16 = (_Float16*)(ws);                       // 512*256*2 = 256 KB
    float*    kf32 = (float*)(ws + 262144);                 // 512*256*4 = 512 KB
    float*    qdp  = (float*)(ws + 262144 + 524288);        // 512*128*4 = 256 KB
    float*    kdm  = (float*)(ws + 262144 + 524288 + 262144);

    ln_proj_kernel<<<NN, 256, 0, stream>>>(node, ln_w, ln_b, proj_w, proj_b,
                                           o_w, o_b, qf16, kf32, qdp, kdm);

    const size_t smem = (size_t)EE * HH * sizeof(_Float16); // 64 KB
    edge_wmma_kernel<<<NN, 256, smem, stream>>>(qf16, kf32, o_w, qdp, kdm, out);
}
